// AxialAttention_72258529788469
// MI455X (gfx1250) — matrix operand order repaired and checked by reading, not compiled, not run on hardware
//
#include <hip/hip_runtime.h>
#include <stdint.h>

// ---------------------------------------------------------------------------
// Types for CDNA5 WMMA (wave32): bf16 A/B fragments (16 halves = 8 VGPRs),
// f32 C/D accumulator (8 VGPRs).
// ---------------------------------------------------------------------------
typedef __attribute__((ext_vector_type(16))) __bf16 bf16x16;
typedef __attribute__((ext_vector_type(8)))  float  v8f;

union Frag {
  bf16x16      v;
  unsigned int u[8];
};

// TDM descriptor vector types (this toolchain exposes the 6-arg builtin:
// (uint32x4 g0, int32x8 g1, int32x4, int32x4, int32x8, i32 cpol))
typedef unsigned int u32x4 __attribute__((ext_vector_type(4)));
typedef int          i32x8 __attribute__((ext_vector_type(8)));
typedef int          i32x4 __attribute__((ext_vector_type(4)));

#if __has_builtin(__builtin_amdgcn_tensor_load_to_lds) && \
    __has_builtin(__builtin_amdgcn_s_wait_tensorcnt)
#define USE_TDM 1
#else
#define USE_TDM 0
#endif

__device__ __forceinline__ unsigned short f32_to_bf16_rne(float f) {
  unsigned int u = __float_as_uint(f);
  u += 0x7fffu + ((u >> 16) & 1u);
  return (unsigned short)(u >> 16);
}

__device__ __forceinline__ unsigned int pack_bf16x2(float a, float b) {
  return (unsigned int)f32_to_bf16_rne(a) | ((unsigned int)f32_to_bf16_rne(b) << 16);
}

// ---------------------------------------------------------------------------
// f32 -> bf16 conversion, 4 elements / thread (all sizes are multiples of 1024)
// ---------------------------------------------------------------------------
__global__ __launch_bounds__(256) void cvt_f32_bf16(const float* __restrict__ in,
                                                    unsigned int* __restrict__ out,
                                                    int n) {
  int i4 = (blockIdx.x * 256 + threadIdx.x) * 4;
  if (i4 < n) {
    float4 f = *(const float4*)(in + i4);
    uint2 r;
    r.x = pack_bf16x2(f.x, f.y);
    r.y = pack_bf16x2(f.z, f.w);
    *(uint2*)(out + (i4 >> 1)) = r;
  }
}

// ---------------------------------------------------------------------------
// bf16 WMMA GEMM: C[M,N] = A[M,K] * B[K,N], A/B row-major bf16.
// Block tile 128x128, K-step 64, 256 threads = 8 waves, wave tile 64x32
// (4x2 fragments of 16x16). mode: 0 = bf16 store, 1 = f32 store, 2 = f32 +=.
// A tile is staged by the Tensor Data Mover (2D tile descriptor with LDS
// padding reproducing the 72-element pitch); B tile is staged transposed
// by the waves (TDM cannot transpose).
// ---------------------------------------------------------------------------
#define GEMM_BM 128
#define GEMM_BN 128
#define GEMM_BK 64
#define PITCH_H 72   // padded LDS row pitch in bf16 elements (avoids bank conflicts)
#define PITCH_U 36   // same pitch in 32-bit words

__global__ __launch_bounds__(256)
void gemm_bf16_wmma(const unsigned short* __restrict__ A,
                    const unsigned short* __restrict__ B,
                    unsigned short* __restrict__ Cb,
                    float* __restrict__ Cf,
                    int M, int N, int K, int mode) {
  // NOTE: sA is declared first so its LDS offset is 0 (used as D#.lds_addr).
  __shared__ __align__(16) unsigned int   sA[GEMM_BM * PITCH_U];   // A tile row-major
  __shared__ __align__(16) unsigned short sBt[GEMM_BN * PITCH_H];  // B tile transposed [N][K]

  const int tid  = threadIdx.x;
  const int wave = tid >> 5;
  const int lane = tid & 31;
  const int wm   = wave >> 2;          // 0..1 (M)
  const int wn   = wave & 3;           // 0..3 (N)
  const int l15  = lane & 15;
  const int lh   = lane >> 4;          // lane half
  const int m0   = blockIdx.y * GEMM_BM;
  const int n0   = blockIdx.x * GEMM_BN;

  const v8f vzero = {};
  v8f acc[4][2];
#pragma unroll
  for (int mi = 0; mi < 4; ++mi)
#pragma unroll
    for (int ni = 0; ni < 2; ++ni) acc[mi][ni] = vzero;

#if USE_TDM
  // D# group1 (invariant over the K loop): data_size=2B (code 1), pad_enable,
  // pad_interval = 32 dwords (one 64-elem row, code 4), pad_amount = 4 dwords
  // (16B -> 72-elem pitch, code 3); tensor dims (K x M), tile 64x128, stride K.
  const int g1d0 = (1 << 16) | (1 << 20) | (4 << 22) | (3 << 25);
  const i32x8 dgrp1 = { g1d0,
                        (K & 0xffff) << 16,                  // tensor_dim0 lo
                        (K >> 16) | ((M & 0xffff) << 16),    // dim0 hi | dim1 lo
                        (M >> 16) | (GEMM_BK << 16),         // dim1 hi | tile_dim0
                        GEMM_BM,                             // tile_dim1 (tile_dim2 = 0)
                        K,                                   // tensor_dim0_stride lo
                        0, 0 };
  const i32x4 z4 = { 0, 0, 0, 0 };
  const i32x8 z8 = { 0, 0, 0, 0, 0, 0, 0, 0 };
#endif

  for (int k0 = 0; k0 < K; k0 += GEMM_BK) {
#if USE_TDM
    // ---- stage A tile (128x64 bf16) via Tensor Data Mover, wave 0 only ----
    if (wave == 0) {
      unsigned long long ga =
          (unsigned long long)(const void*)(A + ((size_t)m0 * K + k0));
      u32x4 dgrp0;
      dgrp0.x = 1u;                                   // count = 1, user mode
      dgrp0.y = 0u;                                   // lds_addr = offset of sA
      dgrp0.z = (unsigned)(ga & 0xffffffffull);       // global_addr[31:0]
      dgrp0.w = (unsigned)(ga >> 32) | (2u << 30);    // global_addr[56:32] | type=2
      __builtin_amdgcn_tensor_load_to_lds(dgrp0, dgrp1, z4, z4, z8, 0);
      __builtin_amdgcn_s_wait_tensorcnt(0);
    }
#else
    // ---- stage A tile (128x64 bf16): 256 thr x 4 chunks of 8 bf16 ----
#pragma unroll
    for (int i = 0; i < 4; ++i) {
      int c    = tid + 256 * i;
      int row  = c >> 3;
      int koff = (c & 7) << 3;
      uint4 val = *(const uint4*)(A + (size_t)(m0 + row) * K + (k0 + koff));
      unsigned int* dst = &sA[row * PITCH_U + (koff >> 1)];
      dst[0] = val.x; dst[1] = val.y; dst[2] = val.z; dst[3] = val.w;
    }
#endif
    // ---- stage B tile (64x128 bf16) transposed into LDS ----
#pragma unroll
    for (int i = 0; i < 4; ++i) {
      int c    = tid + 256 * i;
      int erow = c >> 4;
      int noff = (c & 15) << 3;
      uint4 val = *(const uint4*)(B + (size_t)(k0 + erow) * N + (n0 + noff));
      const unsigned short* pv = (const unsigned short*)&val;
#pragma unroll
      for (int j = 0; j < 8; ++j)
        sBt[(noff + j) * PITCH_H + erow] = pv[j];
    }
    if (k0 + GEMM_BK < K) {  // hint next tiles into cache (global_prefetch_b8)
      __builtin_prefetch(A + (size_t)(m0 + (tid >> 3)) * K + (k0 + GEMM_BK), 0, 0);
      __builtin_prefetch(B + (size_t)(k0 + GEMM_BK + (tid >> 4)) * N + n0, 0, 0);
    }
    __syncthreads();

    const unsigned int* sBu = (const unsigned int*)sBt;
#pragma unroll
    for (int kk = 0; kk < GEMM_BK; kk += 32) {
      Frag fa[4], fb[2];
#pragma unroll
      for (int mi = 0; mi < 4; ++mi) {
        // A frag: row = lane&15; K = 16 contiguous from kk + lh*16
        const unsigned int* p = &sA[(wm * 64 + mi * 16 + l15) * PITCH_U + ((kk + lh * 16) >> 1)];
#pragma unroll
        for (int j = 0; j < 4; ++j) fa[mi].u[j] = p[j];
#pragma unroll
        for (int j = 0; j < 4; ++j) fa[mi].u[4 + j] = p[4 + j];
      }
#pragma unroll
      for (int ni = 0; ni < 2; ++ni) {
        // B frag: col = lane&15; K = 16 contiguous from kk + lh*16 (Bt row-major in K)
        const unsigned int* p = &sBu[(wn * 32 + ni * 16 + l15) * PITCH_U + ((kk + lh * 16) >> 1)];
#pragma unroll
        for (int j = 0; j < 8; ++j) fb[ni].u[j] = p[j];
      }
#pragma unroll
      for (int mi = 0; mi < 4; ++mi)
#pragma unroll
        for (int ni = 0; ni < 2; ++ni)
          acc[mi][ni] = __builtin_amdgcn_wmma_f32_16x16x32_bf16(
              false, fa[mi].v, false, fb[ni].v, (short)0, acc[mi][ni], false, false);
    }
    __syncthreads();
  }

  // ---- epilogue: C frag row = e + 8*lh, col = lane&15 ----
#pragma unroll
  for (int mi = 0; mi < 4; ++mi) {
    int rbase = m0 + wm * 64 + mi * 16 + lh * 8;
#pragma unroll
    for (int ni = 0; ni < 2; ++ni) {
      int col = n0 + wn * 32 + ni * 16 + l15;
#pragma unroll
      for (int e = 0; e < 8; ++e) {
        size_t idx = (size_t)(rbase + e) * N + col;
        float val  = acc[mi][ni][e];
        if (mode == 0)      Cb[idx] = f32_to_bf16_rne(val);
        else if (mode == 1) Cf[idx] = val;
        else                Cf[idx] += val;
      }
    }
  }
}

// ---------------------------------------------------------------------------
// Per-head linear attention (no softmax): O = (Q K^T * d^-1/2) V, t=64, dh=64.
// One workgroup (4 waves) per (group, head); each wave owns a 16-row strip.
// QKV buffer is natural token order (T, 3072); axis-0 uses stride-64 rows.
// ---------------------------------------------------------------------------
__global__ __launch_bounds__(128)
void axial_attn(const unsigned short* __restrict__ qkv,
                unsigned short* __restrict__ att,
                int axis) {
  __shared__ __align__(16) unsigned int   qs[64 * PITCH_U];   // Q rows [t][dh]
  __shared__ __align__(16) unsigned int   ks[64 * PITCH_U];   // K rows [t][dh]
  __shared__ __align__(16) unsigned short vt[64 * PITCH_H];   // V transposed [dh][t]
  __shared__ __align__(16) unsigned short sS[64 * PITCH_H];   // bf16 scores [t][t]

  const int g    = blockIdx.x >> 4;
  const int head = blockIdx.x & 15;
  int T0, strideT;
  if (axis == 0) { int b = g >> 6, w = g & 63; T0 = b * 4096 + w; strideT = 64; }
  else           { T0 = g * 64; strideT = 1; }

  const int tid  = threadIdx.x;
  const int wave = tid >> 5;
  const int lane = tid & 31;
  const int l15  = lane & 15;
  const int lh   = lane >> 4;
  const int qcol = head * 64;
  const int kcol = 1024 + head * 64;
  const int vcol = 2048 + head * 64;

  // ---- load Q, K (row-major) and V (transposed) tiles into LDS ----
#pragma unroll
  for (int i = 0; i < 4; ++i) {
    int c    = tid + 128 * i;
    int t    = c >> 3;
    int koff = (c & 7) << 3;
    size_t rb = (size_t)(T0 + t * strideT) * 3072;
    uint4 qv = *(const uint4*)(qkv + rb + qcol + koff);
    uint4 kv = *(const uint4*)(qkv + rb + kcol + koff);
    uint4 vv = *(const uint4*)(qkv + rb + vcol + koff);
    unsigned int* qd = &qs[t * PITCH_U + (koff >> 1)];
    qd[0] = qv.x; qd[1] = qv.y; qd[2] = qv.z; qd[3] = qv.w;
    unsigned int* kd = &ks[t * PITCH_U + (koff >> 1)];
    kd[0] = kv.x; kd[1] = kv.y; kd[2] = kv.z; kd[3] = kv.w;
    const unsigned short* pv = (const unsigned short*)&vv;
#pragma unroll
    for (int j = 0; j < 8; ++j) vt[(koff + j) * PITCH_H + t] = pv[j];
  }
  __syncthreads();

  const v8f vzero = {};

  // ---- phase 1: S strip (16x64) = Q_strip (16x64) x K^T ----
  v8f sacc[4];
#pragma unroll
  for (int nf = 0; nf < 4; ++nf) sacc[nf] = vzero;
#pragma unroll
  for (int kb = 0; kb < 64; kb += 32) {
    Frag fa;
    const unsigned int* pa = &qs[(wave * 16 + l15) * PITCH_U + ((kb + lh * 16) >> 1)];
#pragma unroll
    for (int j = 0; j < 4; ++j) fa.u[j] = pa[j];
#pragma unroll
    for (int j = 0; j < 4; ++j) fa.u[4 + j] = pa[4 + j];
#pragma unroll
    for (int nf = 0; nf < 4; ++nf) {
      Frag fb;  // B[e][j] = K[j][e]: read K row j, contiguous e
      const unsigned int* pb = &ks[(nf * 16 + l15) * PITCH_U + ((kb + lh * 16) >> 1)];
#pragma unroll
      for (int j = 0; j < 8; ++j) fb.u[j] = pb[j];
      sacc[nf] = __builtin_amdgcn_wmma_f32_16x16x32_bf16(
          false, fa.v, false, fb.v, (short)0, sacc[nf], false, false);
    }
  }
  // scale by d^-1/2 = 1/32 (full dim, per reference) and round to bf16
#pragma unroll
  for (int nf = 0; nf < 4; ++nf)
#pragma unroll
    for (int e = 0; e < 8; ++e)
      sS[(wave * 16 + e + lh * 8) * PITCH_H + nf * 16 + l15] =
          f32_to_bf16_rne(sacc[nf][e] * 0.03125f);
  __syncthreads();

  // ---- phase 2: O strip (16x64) = S_strip (16x64) x V (64x64) ----
  v8f oacc[4];
#pragma unroll
  for (int nf = 0; nf < 4; ++nf) oacc[nf] = vzero;
  const unsigned int* sSu = (const unsigned int*)sS;
  const unsigned int* vtu = (const unsigned int*)vt;
#pragma unroll
  for (int kb = 0; kb < 64; kb += 32) {
    Frag fa;
    const unsigned int* pa = &sSu[(wave * 16 + l15) * PITCH_U + ((kb + lh * 16) >> 1)];
#pragma unroll
    for (int j = 0; j < 4; ++j) fa.u[j] = pa[j];
#pragma unroll
    for (int j = 0; j < 4; ++j) fa.u[4 + j] = pa[4 + j];
#pragma unroll
    for (int nf = 0; nf < 4; ++nf) {
      Frag fb;  // B[j][n] = V[j][n] = vt[n][j]: read vt row n, contiguous j
      const unsigned int* pb = &vtu[(nf * 16 + l15) * PITCH_U + ((kb + lh * 16) >> 1)];
#pragma unroll
      for (int j = 0; j < 8; ++j) fb.u[j] = pb[j];
      oacc[nf] = __builtin_amdgcn_wmma_f32_16x16x32_bf16(
          false, fa.v, false, fb.v, (short)0, oacc[nf], false, false);
    }
  }

  // ---- write O strip to attention buffer (natural token order, bf16) ----
#pragma unroll
  for (int nf = 0; nf < 4; ++nf)
#pragma unroll
    for (int e = 0; e < 8; ++e) {
      int t = wave * 16 + e + lh * 8;
      size_t idx = (size_t)(T0 + t * strideT) * 1024 + head * 64 + nf * 16 + l15;
      att[idx] = f32_to_bf16_rne(oacc[nf][e]);
    }
}

// ---------------------------------------------------------------------------
// Host-side pipeline: cvt -> [per axis] cvt weights, QKV GEMM, attention,
// out-proj GEMM (write then accumulate into d_out).
// ---------------------------------------------------------------------------
extern "C" void kernel_launch(void* const* d_in, const int* in_sizes, int n_in,
                              void* d_out, int out_size, void* d_ws, size_t ws_size,
                              hipStream_t stream) {
  const float* x     = (const float*)d_in[0];
  const float* wq[2] = {(const float*)d_in[1], (const float*)d_in[3]};
  const float* wo[2] = {(const float*)d_in[2], (const float*)d_in[4]};
  float* out = (float*)d_out;

  const int Mtok = 32768;  // 8*64*64 tokens
  const int D    = 1024;
  const int N3   = 3072;

  char* p = (char*)d_ws;
  auto take = [&](size_t bytes) {
    char* r = p;
    p += (bytes + 255) & ~(size_t)255;
    return r;
  };
  unsigned short* xb   = (unsigned short*)take((size_t)Mtok * D * 2);   //  64 MB
  unsigned short* qkvb = (unsigned short*)take((size_t)Mtok * N3 * 2);  // 192 MB
  unsigned short* attb = (unsigned short*)take((size_t)Mtok * D * 2);   //  64 MB
  unsigned short* wqb  = (unsigned short*)take((size_t)D * N3 * 2);     //   6 MB
  unsigned short* wob  = (unsigned short*)take((size_t)D * D * 2);      //   2 MB

  // x -> bf16 (once)
  {
    int n = Mtok * D;
    cvt_f32_bf16<<<n / 1024, 256, 0, stream>>>(x, (unsigned int*)xb, n);
  }

  for (int axis = 0; axis < 2; ++axis) {
    int nq = D * N3;
    cvt_f32_bf16<<<nq / 1024, 256, 0, stream>>>(wq[axis], (unsigned int*)wqb, nq);
    int no = D * D;
    cvt_f32_bf16<<<no / 1024, 256, 0, stream>>>(wo[axis], (unsigned int*)wob, no);

    // QKV projection: (32768 x 1024) x (1024 x 3072) -> bf16
    dim3 g1(N3 / GEMM_BN, Mtok / GEMM_BM);
    gemm_bf16_wmma<<<g1, 256, 0, stream>>>(xb, wqb, qkvb, nullptr, Mtok, N3, D, 0);

    // per-head attention: 512 groups x 16 heads
    axial_attn<<<512 * 16, 128, 0, stream>>>(qkvb, attb, axis);

    // output projection: (32768 x 1024) x (1024 x 1024) -> f32 (write / accumulate)
    dim3 g2(D / GEMM_BN, Mtok / GEMM_BM);
    gemm_bf16_wmma<<<g2, 256, 0, stream>>>(attb, wob, nullptr, out, Mtok, D, D,
                                           axis == 0 ? 1 : 2);
  }
}
